// LSM_partition_12627203850875
// MI455X (gfx1250) — matrix operand order repaired and checked by reading, not compiled, not run on hardware
//
#include <hip/hip_runtime.h>

// ---------------- problem constants ----------------
#define TSTEPS 512
#define BATCH  64
#define INSZ   256
#define NN     1024
#define PARTS  4

// ---------------- workspace layout (bytes) ----------------
#define OFF_CURR  (0ull)                                   // f32 [T*B, N]      = 134217728
#define OFF_XBF   (134217728ull)                           // bf16 [T*B, INSZ]  =  16777216
#define OFF_WINS  (150994944ull)                           // bf16 [P, N, INSZ] =   2097152
#define OFF_WREC  (153092096ull)                           // bf16 [N, N]       =   2097152
#define OFF_SPK   (155189248ull)                           // bf16 2*[B, N]     =    262144
#define OFF_BAR   (155451392ull)                           // barrier counters  =       256

#define SPK_BYTES (BATCH * NN * 2)                         // 131072 per buffer

typedef __attribute__((ext_vector_type(4)))  unsigned int v4u;
typedef __attribute__((ext_vector_type(8)))  unsigned int v8u;
typedef __attribute__((ext_vector_type(16))) __bf16      v16bf;
typedef __attribute__((ext_vector_type(8)))  float       v8f;

__device__ __forceinline__ unsigned short f2bf(float f) {
  unsigned u = __builtin_bit_cast(unsigned, f);
  u += 0x7fffu + ((u >> 16) & 1u);          // round-to-nearest-even
  return (unsigned short)(u >> 16);
}

// A fragment: 16x32 bf16, row-major source, ldk in halves.
// ISA layout: lanes 0-15 row M=lane, K in {0..7}U{16..23}; lanes 16-31 same rows, K+8.
__device__ __forceinline__ v16bf load_a_frag(const unsigned short* tile, int ldk, int lane) {
  const unsigned short* p = tile + (lane & 15) * ldk + ((lane >> 4) << 3);
  v4u lo = *(const v4u*)(p);        // K base+0..7   -> VGPR0..3
  v4u hi = *(const v4u*)(p + 16);   // K base+16..23 -> VGPR4..7
  v8u r;
  r[0] = lo[0]; r[1] = lo[1]; r[2] = lo[2]; r[3] = lo[3];
  r[4] = hi[0]; r[5] = hi[1]; r[6] = hi[2]; r[7] = hi[3];
  return __builtin_bit_cast(v16bf, r);
}

// B fragment: 32x16 bf16 where B[k][n] is stored K-contiguous per column
// (i.e. source is row-major W[n][k]).  Lane = column (n0+lane&15),
// lanes 0-15 hold K {0..7}U{16..23}, lanes 16-31 hold K {8..15}U{24..31} (matches A).
__device__ __forceinline__ v16bf load_b_frag(const unsigned short* tile, int ldk, int lane) {
  const unsigned short* p = tile + (lane & 15) * ldk + ((lane >> 4) << 3);
  v4u lo = *(const v4u*)(p);
  v4u hi = *(const v4u*)(p + 16);
  v8u r;
  r[0] = lo[0]; r[1] = lo[1]; r[2] = lo[2]; r[3] = lo[3];
  r[4] = hi[0]; r[5] = hi[1]; r[6] = hi[2]; r[7] = hi[3];
  return __builtin_bit_cast(v16bf, r);
}

// ---------------- f32 -> bf16 conversion ----------------
__global__ void cvt_f32_bf16(const float* __restrict__ in, unsigned short* __restrict__ out, int n) {
  int i = blockIdx.x * blockDim.x + threadIdx.x;
  int stride = gridDim.x * blockDim.x;
  for (; i < n; i += stride) out[i] = f2bf(in[i]);
}

// ---------------- phase 1: input projection GEMM ----------------
// curr[T*B, N] = x_bf @ Wins[p]^T + b1 ; each wave computes a 16x64 output strip.
__global__ void __launch_bounds__(128) proj_kernel(
    const unsigned short* __restrict__ xbf,     // [T*B, INSZ]
    const unsigned short* __restrict__ winsbf,  // [P, N, INSZ]
    const float* __restrict__ b1,               // [N]
    float* __restrict__ curr) {                 // [T*B, N]
  const int lane = threadIdx.x & 31;
  const int wave = threadIdx.x >> 5;
  const int gw   = blockIdx.x * 4 + wave;       // 0 .. 32767
  const int mt   = gw >> 4;                     // 2048 M tiles of 16 rows
  const int n0   = (gw & 15) * 64;              // 16 strips of 64 cols
  const int p    = mt >> 9;                     // 512 M-tiles per partition
  const unsigned short* xrow = xbf + (size_t)mt * 16 * INSZ;
  const unsigned short* wb   = winsbf + (size_t)p * NN * INSZ;

  v8f acc[4] = {};
  for (int k0 = 0; k0 < INSZ; k0 += 32) {
    v16bf a = load_a_frag(xrow + k0, INSZ, lane);
#pragma unroll
    for (int nt = 0; nt < 4; ++nt) {
      v16bf b = load_b_frag(wb + (size_t)(n0 + nt * 16) * INSZ + k0, INSZ, lane);
      acc[nt] = __builtin_amdgcn_wmma_f32_16x16x32_bf16(
          false, a, false, b, (short)0, acc[nt], false, false);
    }
  }
  const int col = lane & 15;
  const int mo  = (lane >> 4) << 3;             // hi half-wave -> rows +8
#pragma unroll
  for (int nt = 0; nt < 4; ++nt) {
    int n = n0 + nt * 16 + col;
    float bias = b1[n];
#pragma unroll
    for (int r = 0; r < 8; ++r) {
      curr[(size_t)(mt * 16 + mo + r) * NN + n] = acc[nt][r] + bias;
    }
  }
}

// ---------------- phase 2: persistent LIF scan ----------------
// 16 WGs x 4 waves; wave gw owns output columns [gw*16, gw*16+16) for all 64 batch rows.
// syn/mem state lives in VGPRs for all 512 steps.  Each step the WG async-copies the
// 128 KB spike buffer (global, double-buffered) into LDS once; A-fragments then come
// from LDS (ds_load_b128), cutting redundant L2 reads 4x.  One device-wide atomic
// barrier per step.
__global__ void __launch_bounds__(128) scan_kernel(
    const float* __restrict__ curr,             // [T*B, N]
    const unsigned short* __restrict__ wrecbf,  // [N, N] row-major (n, k)
    const float* __restrict__ brec,             // [N]
    unsigned short* __restrict__ spkbuf,        // 2 * [B, N] bf16, pre-zeroed
    float* __restrict__ out,                    // [T, B, N]
    unsigned int* __restrict__ bar) {           // bar[0]=count, bar[1]=generation
  extern __shared__ char smem[];                // 131072 B: spike tile [B, N] bf16
  unsigned short* spk_lds = (unsigned short*)smem;

  const int lane = threadIdx.x & 31;
  const int wave = threadIdx.x >> 5;
  const int gw   = blockIdx.x * 4 + wave;       // 0..63
  const int n0   = gw * 16;
  const int col  = n0 + (lane & 15);
  const int mo   = (lane >> 4) << 3;
  const float brv = brec[col];
  const unsigned short* wtile = wrecbf + (size_t)n0 * NN;
  const unsigned lds_base = (unsigned)(uintptr_t)spk_lds;  // low 32 bits = LDS byte addr

  float syn[4][8], mem[4][8];
#pragma unroll
  for (int i = 0; i < 4; ++i)
#pragma unroll
    for (int r = 0; r < 8; ++r) { syn[i][r] = 0.0f; mem[i][r] = 0.0f; }

  for (int t = 0; t < TSTEPS; ++t) {
    const unsigned short* spkA = spkbuf + (size_t)(t & 1) * (BATCH * NN);
    unsigned short*       spkB = spkbuf + (size_t)((t + 1) & 1) * (BATCH * NN);

    // ---- async copy of full spike buffer (131072 B) global -> LDS ----
    {
      const char* g = (const char*)spkA;
      const int tid = threadIdx.x;
#pragma unroll 8
      for (int i = 0; i < 64; ++i) {            // 128 lanes x 64 iters x 16 B = 128 KB
        int off = (i * 128 + tid) * 16;
        unsigned laddr = lds_base + (unsigned)off;
        const char* ga = g + off;
        asm volatile("global_load_async_to_lds_b128 %0, %1, off"
                     :: "v"(laddr), "v"(ga) : "memory");
      }
      asm volatile("s_wait_asynccnt 0x0" ::: "memory");
      __syncthreads();
    }

    // ---- dot = spk @ Wrec^T restricted to our 16 columns (A from LDS, B from L2) ----
    v8f acc[4] = {};
    for (int k0 = 0; k0 < NN; k0 += 32) {
      v16bf b = load_b_frag(wtile + k0, NN, lane);
#pragma unroll
      for (int m = 0; m < 4; ++m) {
        v16bf a = load_a_frag(spk_lds + (size_t)m * 16 * NN + k0, NN, lane);
        acc[m] = __builtin_amdgcn_wmma_f32_16x16x32_bf16(
            false, a, false, b, (short)0, acc[m], false, false);
      }
    }

    // ---- LIF state update (state held in registers across all steps) ----
    const float* currt = curr + (size_t)t * BATCH * NN;
#pragma unroll
    for (int m = 0; m < 4; ++m) {
#pragma unroll
      for (int r = 0; r < 8; ++r) {
        int bb = m * 16 + mo + r;
        float c     = currt[(size_t)bb * NN + col];
        float syn_n = 0.9f * syn[m][r] + c + acc[m][r] + brv;
        float reset = (mem[m][r] > 20.0f) ? 20.0f : 0.0f;    // from *previous* mem
        float mem_n = 0.9f * mem[m][r] + syn_n - reset;
        float s     = (mem_n > 20.0f) ? 1.0f : 0.0f;
        syn[m][r] = syn_n;
        mem[m][r] = mem_n;
        out[(size_t)t * BATCH * NN + (size_t)bb * NN + col] = s;
        spkB[(size_t)bb * NN + col] = (s > 0.0f) ? (unsigned short)0x3F80u
                                                 : (unsigned short)0u;
      }
    }

    if (t + 1 < TSTEPS) {  // warm L2 for next step's current slice
      __builtin_prefetch(currt + BATCH * NN + (size_t)mo * NN + col, 0, 0);
    }

    // ---- device-wide barrier: spkB visible before anyone reads it as spkA;
    //      also protects spk_lds from being overwritten while still in use ----
    __syncthreads();
    if (threadIdx.x == 0) {
      __threadfence();
      unsigned prev = __hip_atomic_fetch_add(&bar[0], 1u, __ATOMIC_ACQ_REL,
                                             __HIP_MEMORY_SCOPE_AGENT);
      if (prev == (unsigned)(gridDim.x - 1)) {
        __hip_atomic_store(&bar[0], 0u, __ATOMIC_RELAXED, __HIP_MEMORY_SCOPE_AGENT);
        __hip_atomic_fetch_add(&bar[1], 1u, __ATOMIC_ACQ_REL, __HIP_MEMORY_SCOPE_AGENT);
      } else {
        while (__hip_atomic_load(&bar[1], __ATOMIC_ACQUIRE, __HIP_MEMORY_SCOPE_AGENT)
               < (unsigned)(t + 1)) {
          __builtin_amdgcn_s_sleep(2);
        }
      }
    }
    __syncthreads();
  }
}

extern "C" void kernel_launch(void* const* d_in, const int* in_sizes, int n_in,
                              void* d_out, int out_size, void* d_ws, size_t ws_size,
                              hipStream_t stream) {
  const float* x    = (const float*)d_in[0];   // [T,B,INSZ]
  const float* Wins = (const float*)d_in[1];   // [P,N,INSZ]
  const float* b1   = (const float*)d_in[2];   // [N]
  const float* Wrec = (const float*)d_in[3];   // [N,N]
  const float* brec = (const float*)d_in[4];   // [N]
  float* out = (float*)d_out;                  // [T,B,N] f32

  char* ws = (char*)d_ws;
  float*          curr   = (float*)(ws + OFF_CURR);
  unsigned short* xbf    = (unsigned short*)(ws + OFF_XBF);
  unsigned short* winsbf = (unsigned short*)(ws + OFF_WINS);
  unsigned short* wrecbf = (unsigned short*)(ws + OFF_WREC);
  unsigned short* spk    = (unsigned short*)(ws + OFF_SPK);
  unsigned int*   bar    = (unsigned int*)(ws + OFF_BAR);

  // zero spike double-buffer + barrier counters (covers OFF_SPK..OFF_BAR+256)
  hipMemsetAsync(ws + OFF_SPK, 0, (OFF_BAR - OFF_SPK) + 256, stream);

  // f32 -> bf16 conversions
  {
    int n = TSTEPS * BATCH * INSZ;
    cvt_f32_bf16<<<(n + 2047) / 2048, 256, 0, stream>>>(x, xbf, n);
  }
  {
    int n = PARTS * NN * INSZ;
    cvt_f32_bf16<<<(n + 2047) / 2048, 256, 0, stream>>>(Wins, winsbf, n);
  }
  {
    int n = NN * NN;
    cvt_f32_bf16<<<(n + 2047) / 2048, 256, 0, stream>>>(Wrec, wrecbf, n);
  }

  // phase 1: input projection (2048 M-tiles x 16 strips, 4 waves/WG)
  proj_kernel<<<8192, 128, 0, stream>>>(xbf, winsbf, b1, curr);

  // phase 2: persistent recurrent scan (16 WGs, 128 KB dynamic LDS spike tile each)
  scan_kernel<<<16, 128, SPK_BYTES, stream>>>(curr, wrecbf, brec, spk, out, bar);
}
